// FrameTransformerDecoder_18210661335566
// MI455X (gfx1250) — compile-verified, matrix operand order checked by reading, not run on hardware
//
#include <hip/hip_runtime.h>
#include <hip/hip_bf16.h>
#include <math.h>

// ---------------------------------------------------------------------------
// FrameTransformerDecoder forward for MI455X (gfx1250), wave32 + WMMA bf16.
// GEMM path: TDM tensor_load_to_lds (f32 tiles) -> LDS convert to bf16 ->
// ds_load fragments -> v_wmma_f32_16x16x32_bf16.  All GEMM dims in this net
// are multiples of the 128x128x32 tile, so no bounds checks anywhere.
// ---------------------------------------------------------------------------

typedef __attribute__((ext_vector_type(16))) __bf16 v16bf;
typedef __attribute__((ext_vector_type(8)))  float  v8f;

#define Bc   4
#define CHc  8
#define NBc  4
#define Cc   1024     // BINS
#define Wc   1024     // CS (frames)
#define FFc  2048
#define HDc  256      // Cc / NBc
#define BWc  4096     // Bc * Wc
#define EPSc 1e-5f

#if __has_builtin(__builtin_amdgcn_tensor_load_to_lds) && \
    __has_builtin(__builtin_amdgcn_s_wait_tensorcnt)
#define USE_TDM 1
#else
#define USE_TDM 0
#endif

// round-to-nearest-even f32 -> bf16, packed pair (a = low 16, b = high 16)
__device__ __forceinline__ unsigned pack_bf16(float a, float b) {
  unsigned ua = __float_as_uint(a), ub = __float_as_uint(b);
  ua += 0x7FFFu + ((ua >> 16) & 1u);
  ub += 0x7FFFu + ((ub >> 16) & 1u);
  return (ua >> 16) | (ub & 0xFFFF0000u);
}

union FragU { unsigned u[8]; v16bf v; };

#if USE_TDM
typedef __attribute__((ext_vector_type(4))) unsigned u32x4;
typedef __attribute__((ext_vector_type(8))) int      i32x8;
typedef __attribute__((ext_vector_type(4))) int      i32x4;

// generic LDS pointer: hw uses addr[31:0] as the wave-relative LDS address
__device__ __forceinline__ unsigned lds_addr32(const void* p) {
  return (unsigned)(unsigned long long)p;
}

// 2D TDM tile load: tile_d0 contiguous elems (f32) x tile_d1 rows, row stride
// ld_elems (f32 units). D# bit layout per CDNA5 ISA ch.8.
__device__ __forceinline__ void tdm_load_tile(unsigned lds_byte,
                                              const float* gptr,
                                              long ld_elems,
                                              int tile_d0, int tile_d1)
{
  unsigned long long ga = (unsigned long long)(const void*)gptr;
  u32x4 g0 = {};
  g0[0] = 1u;                                            // count=1, user mode
  g0[1] = lds_byte;                                      // lds_addr (bytes)
  g0[2] = (unsigned)(ga & 0xFFFFFFFFu);                  // global_addr[31:0]
  g0[3] = (unsigned)((ga >> 32) & 0x01FFFFFFu)           // global_addr[56:32]
          | (2u << 30);                                  // type = 2 (image)
  const unsigned td0 = 0x40000000u, td1 = 0x40000000u;   // huge dims: no OOB
  unsigned long long st = (unsigned long long)ld_elems;  // dim0 stride (elems)
  i32x8 g1 = {};
  g1[0] = (int)(2u << 16);                               // data_size=2 (4B)
  g1[1] = (int)((td0 & 0xFFFFu) << 16);                  // tensor_dim0 lo16
  g1[2] = (int)((td0 >> 16) | ((td1 & 0xFFFFu) << 16));  // td0 hi | td1 lo
  g1[3] = (int)((td1 >> 16) | ((unsigned)tile_d0 << 16));// td1 hi | tile_dim0
  g1[4] = (int)((unsigned)tile_d1 & 0xFFFFu);            // tile_dim1 (dim2=0)
  g1[5] = (int)(unsigned)(st & 0xFFFFFFFFu);             // dim0_stride lo32
  g1[6] = (int)(unsigned)((st >> 32) & 0xFFFFu);         // dim0_stride hi16
  g1[7] = 0;
  i32x4 z4 = {};
#if defined(__clang_major__) && (__clang_major__ >= 23)
  i32x8 z8 = {};
  __builtin_amdgcn_tensor_load_to_lds(g0, g1, z4, z4, z8, 0);
#else
  __builtin_amdgcn_tensor_load_to_lds(g0, g1, z4, z4, 0);
#endif
}
#endif // USE_TDM

// ---------------------------------------------------------------------------
// Generic batched GEMM:  C = act(alpha * A*B + bias) [+= if beta]
//   A: [M,K] row-major, lda;  per-batch offset bb*aB + bn*aN (bz = bb*nbDiv+bn)
//   B: [K,N] row-major (transB=0) or [N,K] row-major (transB=1)
//   C: [M,N] (storeT=0) or written transposed C[n*ldc+m] (storeT=1)
// REQUIRES: M % 128 == 0, N % 128 == 0, K % 32 == 0 (true for all call sites).
// Tiles: 128x128x32, 256 threads = 8 waves, each wave 32(M) x 64(N).
// ---------------------------------------------------------------------------
__global__ __launch_bounds__(256) void gemm_wmma_kernel(
    const float* __restrict__ A, long lda, long aB, long aN,
    const float* __restrict__ Bm, long ldb, long bB, long bN,
    float* __restrict__ Cm, long ldc, long cB, long cN,
    const float* __restrict__ bias,
    int M, int N, int K, float alpha, int beta, int act,
    int transB, int storeT, int nbDiv)
{
  const int bz = blockIdx.z;
  const int bb = bz / nbDiv, bn = bz % nbDiv;
  A  += (long)bb * aB + (long)bn * aN;
  Bm += (long)bb * bB + (long)bn * bN;
  Cm += (long)bb * cB + (long)bn * cN;

  const int tm = blockIdx.y * 128;
  const int tn = blockIdx.x * 128;

#if USE_TDM
  __shared__ float    Asf[128 * 32];  // f32 staging, TDM destination (16KB)
  __shared__ float    Bsf[128 * 32];  // 16KB
#endif
  __shared__ unsigned As[128][16];    // bf16 pairs, WMMA layout (8KB)
  __shared__ unsigned Bs[128][16];    // 8KB

  const int tid  = threadIdx.x;
  const int wave = tid >> 5, lane = tid & 31;
  const int wm = wave >> 1, wn = wave & 1;     // 4x2 wave grid
  const int half = lane >> 4, l15 = lane & 15;

  v8f acc[2][4];
  #pragma unroll
  for (int i = 0; i < 2; ++i)
    #pragma unroll
    for (int j = 0; j < 4; ++j)
      acc[i][j] = v8f{};

  for (int k0 = 0; k0 < K; k0 += 32) {
    __syncthreads();   // previous iteration's LDS reads complete

#if USE_TDM
    // --- DMA the f32 tiles into LDS via the Tensor Data Mover -------------
    if (wave == 0) {
      tdm_load_tile(lds_addr32(Asf), &A[(long)tm * lda + k0], lda, 32, 128);
      if (!transB)
        tdm_load_tile(lds_addr32(Bsf), &Bm[(long)k0 * ldb + tn], ldb, 128, 32);
      else
        tdm_load_tile(lds_addr32(Bsf), &Bm[(long)tn * ldb + k0], ldb, 32, 128);
      __builtin_amdgcn_s_wait_tensorcnt(0);
    }
    __syncthreads();
    // --- convert f32 staging -> packed bf16 WMMA tiles --------------------
    #pragma unroll
    for (int it = 0; it < 8; ++it) {
      int p = tid + it * 256;           // 0..2047 pair index
      int r = p >> 4, cp = p & 15;
      As[r][cp] = pack_bf16(Asf[r * 32 + cp * 2], Asf[r * 32 + cp * 2 + 1]);
      if (!transB)                      // Bsf layout [k][n]
        Bs[r][cp] = pack_bf16(Bsf[(cp * 2) * 128 + r],
                              Bsf[(cp * 2 + 1) * 128 + r]);
      else                              // Bsf layout [n][k]
        Bs[r][cp] = pack_bf16(Bsf[r * 32 + cp * 2], Bsf[r * 32 + cp * 2 + 1]);
    }
#else
    // --- fallback: direct vectorized staging ------------------------------
    if (k0 + 32 < K)
      __builtin_prefetch(&A[(long)(tm + (tid & 127)) * lda + k0 + 32], 0, 1);
    #pragma unroll
    for (int it = 0; it < 8; ++it) {
      int p = tid + it * 256;
      int r = p >> 4, cp = p & 15;
      const float2 a2 =
          *(const float2*)&A[(long)(tm + r) * lda + k0 + cp * 2];
      As[r][cp] = pack_bf16(a2.x, a2.y);
    }
    #pragma unroll
    for (int it = 0; it < 8; ++it) {
      int p = tid + it * 256;
      int c = p >> 4, kp = p & 15;
      int n = tn + c, k = k0 + kp * 2;
      float x0, x1;
      if (!transB) {
        x0 = Bm[(long)k * ldb + n];
        x1 = Bm[(long)(k + 1) * ldb + n];
      } else {
        const float2 b2 = *(const float2*)&Bm[(long)n * ldb + k];
        x0 = b2.x; x1 = b2.y;
      }
      Bs[c][kp] = pack_bf16(x0, x1);
    }
#endif
    __syncthreads();

    // fragment gather per documented 16-bit A/B VGPR layout:
    // lane (half,l15): VGPR p holds K pair starting at 2p + half*8 + (p>=4?8:0)
    FragU af[2], bf[4];
    #pragma unroll
    for (int i = 0; i < 2; ++i) {
      int row = wm * 32 + i * 16 + l15;
      #pragma unroll
      for (int p = 0; p < 8; ++p)
        af[i].u[p] = As[row][p + half * 4 + (p >= 4 ? 4 : 0)];
    }
    #pragma unroll
    for (int j = 0; j < 4; ++j) {
      int col = wn * 64 + j * 16 + l15;
      #pragma unroll
      for (int p = 0; p < 8; ++p)
        bf[j].u[p] = Bs[col][p + half * 4 + (p >= 4 ? 4 : 0)];
    }

    #pragma unroll
    for (int i = 0; i < 2; ++i)
      #pragma unroll
      for (int j = 0; j < 4; ++j)
        acc[i][j] = __builtin_amdgcn_wmma_f32_16x16x32_bf16(
            false, af[i].v, false, bf[j].v, (short)0, acc[i][j], false, false);
  }

  // epilogue: lane (half,l15), VGPR r -> (m_local = half*8 + r, n_local = l15)
  #pragma unroll
  for (int i = 0; i < 2; ++i) {
    #pragma unroll
    for (int j = 0; j < 4; ++j) {
      const int m0 = tm + wm * 32 + i * 16 + half * 8;
      const int n  = tn + wn * 64 + j * 16 + l15;
      const float bv = bias ? bias[n] : 0.f;
      #pragma unroll
      for (int r = 0; r < 8; ++r) {
        float v = acc[i][j][r] * alpha + bv;
        if (act == 1)      { v = fmaxf(v, 0.f); }
        else if (act == 2) { v = fmaxf(v, 0.f); v *= v; }
        long off = storeT ? ((long)n * ldc + (m0 + r))
                          : ((long)(m0 + r) * ldc + n);
        if (beta) Cm[off] += v; else Cm[off] = v;
      }
    }
  }
}

// out[b,w,f] = sum_c in[b,c,f,w] * wt[c]     (in: [B,CH,BINS,CS])
__global__ __launch_bounds__(256) void chanproj_kernel(
    const float* __restrict__ in, const float* __restrict__ wt,
    float* __restrict__ out, long total)
{
  long i = (long)blockIdx.x * 256 + threadIdx.x;
  if (i >= total) return;
  int f = (int)(i % Cc);
  long rem = i / Cc;
  int w = (int)(rem % Wc);
  int b = (int)(rem / Wc);
  float s = 0.f;
  #pragma unroll
  for (int c = 0; c < CHc; ++c)
    s += in[(((long)b * CHc + c) * Cc + f) * Wc + w] * wt[c];
  out[i] = s;
}

// LayerNorm over last dim; one block per row (dst may alias src).
__global__ __launch_bounds__(256) void ln_kernel(
    const float* __restrict__ src, float* __restrict__ dst,
    const float* __restrict__ g, const float* __restrict__ bta, int dim)
{
  long row = blockIdx.x;
  const float* r = src + row * (long)dim;
  float s = 0.f, ss = 0.f;
  for (int j = threadIdx.x; j < dim; j += 256) { float v = r[j]; s += v; ss += v * v; }
  __shared__ float sh1[256], sh2[256];
  sh1[threadIdx.x] = s; sh2[threadIdx.x] = ss;
  __syncthreads();
  for (int st = 128; st > 0; st >>= 1) {
    if (threadIdx.x < st) { sh1[threadIdx.x] += sh1[threadIdx.x + st];
                            sh2[threadIdx.x] += sh2[threadIdx.x + st]; }
    __syncthreads();
  }
  float mean = sh1[0] / dim;
  float var  = sh2[0] / dim - mean * mean;
  float inv  = rsqrtf(var + EPSc);
  float* o = dst + row * (long)dim;
  for (int j = threadIdx.x; j < dim; j += 256)
    o[j] = (r[j] - mean) * inv * g[j] + bta[j];
}

// Depthwise conv along W of [B,W,C]; ker: [C,1,K]; zero pad.
__global__ __launch_bounds__(256) void dwconv_kernel(
    const float* __restrict__ src, float* __restrict__ dst,
    const float* __restrict__ ker, const float* __restrict__ bias,
    int Cn, int Ktap, int pad, long total)
{
  long i = (long)blockIdx.x * 256 + threadIdx.x;
  if (i >= total) return;
  int c = (int)(i % Cn);
  long rem = i / Cn;
  int w = (int)(rem % Wc);
  long b = rem / Wc;
  float s = bias ? bias[c] : 0.f;
  for (int t = 0; t < Ktap; ++t) {
    int ws = w + t - pad;
    if (ws >= 0 && ws < Wc)
      s += src[((long)b * Wc + ws) * Cn + c] * ker[c * Ktap + t];
  }
  dst[i] = s;
}

// in-place row softmax with pre-scale; one block per row
__global__ __launch_bounds__(256) void softmax_kernel(
    float* __restrict__ buf, int dim, float scale)
{
  long row = blockIdx.x;
  float* r = buf + row * (long)dim;
  __shared__ float sh[256];
  float mx = -INFINITY;
  for (int j = threadIdx.x; j < dim; j += 256) mx = fmaxf(mx, r[j] * scale);
  sh[threadIdx.x] = mx; __syncthreads();
  for (int st = 128; st > 0; st >>= 1) {
    if (threadIdx.x < st) sh[threadIdx.x] = fmaxf(sh[threadIdx.x], sh[threadIdx.x + st]);
    __syncthreads();
  }
  mx = sh[0]; __syncthreads();
  float s = 0.f;
  for (int j = threadIdx.x; j < dim; j += 256) {
    float e = __expf(r[j] * scale - mx);
    r[j] = e; s += e;
  }
  sh[threadIdx.x] = s; __syncthreads();
  for (int st = 128; st > 0; st >>= 1) {
    if (threadIdx.x < st) sh[threadIdx.x] += sh[threadIdx.x + st];
    __syncthreads();
  }
  float inv = 1.f / sh[0];
  for (int j = threadIdx.x; j < dim; j += 256) r[j] *= inv;
}

// dst[r, 0:dimSrc] += src[r, :]   (hL + pad(hR))
__global__ __launch_bounds__(256) void addpad_kernel(
    float* __restrict__ dst, const float* __restrict__ src,
    int dimDst, int dimSrc, long total)
{
  long i = (long)blockIdx.x * 256 + threadIdx.x;
  if (i >= total) return;
  long r = i / dimSrc; int j = (int)(i % dimSrc);
  dst[r * (long)dimDst + j] += src[i];
}

// out[b,0,f,w] = x1[b,w,f]
__global__ __launch_bounds__(256) void transpose_out_kernel(
    const float* __restrict__ x1, float* __restrict__ out, long total)
{
  long i = (long)blockIdx.x * 256 + threadIdx.x;
  if (i >= total) return;
  int w = (int)(i % Wc);
  long rem = i / Wc;
  int f = (int)(rem % Cc);
  long b = rem / Cc;
  out[i] = x1[((long)b * Wc + w) * Cc + f];
}

// ---------------------------------------------------------------------------
// host-side helpers
// ---------------------------------------------------------------------------
static void launch_gemm(hipStream_t s,
                        const float* A, long lda, long aB, long aN,
                        const float* Bm, long ldb, long bB, long bN,
                        float* Cm, long ldc, long cB, long cN,
                        const float* bias, int M, int N, int K,
                        float alpha, int beta, int act, int transB, int storeT,
                        int batches, int nbDiv)
{
  dim3 grid((N + 127) / 128, (M + 127) / 128, batches);
  gemm_wmma_kernel<<<grid, 256, 0, s>>>(A, lda, aB, aN, Bm, ldb, bB, bN,
                                        Cm, ldc, cB, cN, bias, M, N, K,
                                        alpha, beta, act, transB, storeT, nbDiv);
}

static void launch_dwconv(hipStream_t s, const float* src, float* dst,
                          const float* ker, const float* bias,
                          int Cn, int Ktap, int pad)
{
  long total = (long)Bc * Wc * Cn;
  dwconv_kernel<<<(unsigned)((total + 255) / 256), 256, 0, s>>>(
      src, dst, ker, bias, Cn, Ktap, pad, total);
}

struct Att {
  const float *er, *kb, *kcb, *kcw, *kw, *ob, *ow, *qb, *qcb, *qcw, *qw,
              *vb, *vcb, *vcw, *vw;
};

#define IN(i) ((const float*)d_in[(i)])
static Att att_at(void* const* d_in, int base) {
  Att a;
  a.er  = IN(base + 0);  a.kb  = IN(base + 1);  a.kcb = IN(base + 2);
  a.kcw = IN(base + 3);  a.kw  = IN(base + 4);  a.ob  = IN(base + 5);
  a.ow  = IN(base + 6);  a.qb  = IN(base + 7);  a.qcb = IN(base + 8);
  a.qcw = IN(base + 9);  a.qw  = IN(base + 10); a.vb  = IN(base + 11);
  a.vcb = IN(base + 12); a.vcw = IN(base + 13); a.vw  = IN(base + 14);
  return a;
}

// MultibandFrameAttention; accumulates result into x1.
static void run_mba(hipStream_t s, const float* hin, const float* mem, float* x1,
                    float* t0, float* tq, float* tk, float* tv, float* sc,
                    const Att& P)
{
  // q/k/v: pointwise proj (+bias) then depthwise conv K=3 pad=1 (+bias)
  launch_gemm(s, hin, Cc, 0, 0, P.qw, Cc, 0, 0, t0, Cc, 0, 0, P.qb,
              BWc, Cc, Cc, 1.f, 0, 0, 0, 0, 1, 1);
  launch_dwconv(s, t0, tq, P.qcw, P.qcb, Cc, 3, 1);
  launch_gemm(s, mem, Cc, 0, 0, P.kw, Cc, 0, 0, t0, Cc, 0, 0, P.kb,
              BWc, Cc, Cc, 1.f, 0, 0, 0, 0, 1, 1);
  launch_dwconv(s, t0, tk, P.kcw, P.kcb, Cc, 3, 1);
  launch_gemm(s, mem, Cc, 0, 0, P.vw, Cc, 0, 0, t0, Cc, 0, 0, P.vb,
              BWc, Cc, Cc, 1.f, 0, 0, 0, 0, 1, 1);
  launch_dwconv(s, t0, tv, P.vcw, P.vcb, Cc, 3, 1);

  // Music-Transformer skew: bias(i,j) = (Q @ Er)[j,i]
  // scores = (Q @ Er)^T, batched over (b, nb); er: [HD, CS] row-major
  launch_gemm(s, tq, Cc, (long)Wc * Cc, HDc,
              P.er, Wc, 0, 0,
              sc, Wc, (long)NBc * Wc * Wc, (long)Wc * Wc, nullptr,
              Wc, Wc, HDc, 1.f, 0, 0, /*transB*/0, /*storeT*/1,
              Bc * NBc, NBc);
  // scores += Q @ K^T
  launch_gemm(s, tq, Cc, (long)Wc * Cc, HDc,
              tk, Cc, (long)Wc * Cc, HDc,
              sc, Wc, (long)NBc * Wc * Wc, (long)Wc * Wc, nullptr,
              Wc, Wc, HDc, 1.f, 1, 0, /*transB*/1, 0, Bc * NBc, NBc);
  // softmax((S)/sqrt(C)) over last dim
  softmax_kernel<<<Bc * NBc * Wc, 256, 0, s>>>(sc, Wc, 1.f / 32.f);
  // t0[b,w, nb*HD + h] = A @ V
  launch_gemm(s, sc, Wc, (long)NBc * Wc * Wc, (long)Wc * Wc,
              tv, Cc, (long)Wc * Cc, HDc,
              t0, Cc, (long)Wc * Cc, HDc, nullptr,
              Wc, HDc, Wc, 1.f, 0, 0, 0, 0, Bc * NBc, NBc);
  // x1 += O @ ow + ob
  launch_gemm(s, t0, Cc, 0, 0, P.ow, Cc, 0, 0, x1, Cc, 0, 0, P.ob,
              BWc, Cc, Cc, 1.f, 1, 0, 0, 0, 1, 1);
}

// ---------------------------------------------------------------------------
// Input index map (top-level dict order: x, skip, params; params flattened by
// sorted key, attention sub-dicts sorted too):
//  0 x, 1 skip, 2..7 b1..b6, 8..22 ea1, 23..37 ea2, 38..43 g1..g6, 44 in_w,
//  45..59 sa1, 60..74 sa2, 75 skip_w, 76 w1L_dw, 77 w1L_pw, 78 w1R_dw,
//  79 w1R_pw, 80 w2_dw, 81 w2_pw, 82 w3, 83 w4
// ---------------------------------------------------------------------------
extern "C" void kernel_launch(void* const* d_in, const int* in_sizes, int n_in,
                              void* d_out, int out_size, void* d_ws, size_t ws_size,
                              hipStream_t stream)
{
  (void)in_sizes; (void)n_in; (void)out_size; (void)ws_size;
  const float* x    = IN(0);
  const float* skip = IN(1);
  const float* b1 = IN(2); const float* b2 = IN(3); const float* b3 = IN(4);
  const float* b4 = IN(5); const float* b5 = IN(6); const float* b6 = IN(7);
  Att ea1 = att_at(d_in, 8);
  Att ea2 = att_at(d_in, 23);
  const float* g1 = IN(38); const float* g2 = IN(39); const float* g3 = IN(40);
  const float* g4 = IN(41); const float* g5 = IN(42); const float* g6 = IN(43);
  const float* in_w = IN(44);
  Att sa1 = att_at(d_in, 45);
  Att sa2 = att_at(d_in, 60);
  const float* skip_w = IN(75);
  const float* w1L_dw = IN(76); const float* w1L_pw = IN(77);
  const float* w1R_dw = IN(78); const float* w1R_pw = IN(79);
  const float* w2_dw  = IN(80); const float* w2_pw  = IN(81);
  const float* w3     = IN(82); const float* w4     = IN(83);
  float* out = (float*)d_out;

  // workspace layout (floats): needs 16*SZ = 64M floats = 256 MB
  const long SZ = (long)Bc * Wc * Cc;     // 4M elements
  float* ws = (float*)d_ws;
  float* xn = ws + 0 * SZ;
  float* sk = ws + 1 * SZ;
  float* x1 = ws + 2 * SZ;
  float* h  = ws + 3 * SZ;
  float* t0 = ws + 4 * SZ;
  float* tq = ws + 5 * SZ;
  float* tk = ws + 6 * SZ;
  float* tv = ws + 7 * SZ;
  float* sc = ws + 8 * SZ;   // 4*SZ  (scores [B,NB,W,W] / FF temps)
  float* ff = ws + 12 * SZ;  // 4*SZ  (hL [B,W,2FF])

  const long tot = SZ;
  const unsigned gEl = (unsigned)((tot + 255) / 256);

  // input / skip channel projections
  chanproj_kernel<<<gEl, 256, 0, stream>>>(x, in_w, xn, tot);
  chanproj_kernel<<<gEl, 256, 0, stream>>>(skip, skip_w, sk, tot);

  // x1 = xn (residual base)
  hipMemcpyAsync(x1, xn, SZ * sizeof(float), hipMemcpyDeviceToDevice, stream);

  // h = LN(xn); x1 += self-attn(h,h) + cross-attn(h,sk)
  ln_kernel<<<BWc, 256, 0, stream>>>(xn, h, g1, b1, Cc);
  run_mba(stream, h, h,  x1, t0, tq, tk, tv, sc, sa1);
  run_mba(stream, h, sk, x1, t0, tq, tk, tv, sc, ea1);

  // FF block 1
  ln_kernel<<<BWc, 256, 0, stream>>>(x1, h, g2, b2, Cc);
  launch_dwconv(stream, h, t0, w1L_dw, nullptr, Cc, 11, 5);
  launch_gemm(stream, t0, Cc, 0, 0, w1L_pw, 2 * FFc, 0, 0, ff, 2 * FFc, 0, 0,
              nullptr, BWc, 2 * FFc, Cc, 1.f, 0, /*relu*/1, 0, 0, 1, 1);
  launch_dwconv(stream, h, t0, w1R_dw, nullptr, Cc, 7, 3);
  launch_gemm(stream, t0, Cc, 0, 0, w1R_pw, FFc, 0, 0, sc, FFc, 0, 0,
              nullptr, BWc, FFc, Cc, 1.f, 0, 0, 0, 0, 1, 1);
  {
    long tp = (long)BWc * FFc;
    addpad_kernel<<<(unsigned)((tp + 255) / 256), 256, 0, stream>>>(
        ff, sc, 2 * FFc, FFc, tp);
  }
  ln_kernel<<<BWc, 256, 0, stream>>>(ff, ff, g3, b3, 2 * FFc);
  launch_dwconv(stream, ff, sc, w2_dw, nullptr, 2 * FFc, 7, 3);
  launch_gemm(stream, sc, 2 * FFc, 0, 0, w2_pw, Cc, 0, 0, x1, Cc, 0, 0,
              nullptr, BWc, Cc, 2 * FFc, 1.f, /*beta*/1, 0, 0, 0, 1, 1);

  // attention blocks 2
  ln_kernel<<<BWc, 256, 0, stream>>>(x1, h, g4, b4, Cc);
  run_mba(stream, h, h,  x1, t0, tq, tk, tv, sc, sa2);
  ln_kernel<<<BWc, 256, 0, stream>>>(x1, h, g5, b5, Cc);
  run_mba(stream, h, sk, x1, t0, tq, tk, tv, sc, ea2);

  // FF block 2: x1 += square(relu(LN(x1) @ w3)) @ w4
  ln_kernel<<<BWc, 256, 0, stream>>>(x1, h, g6, b6, Cc);
  launch_gemm(stream, h, Cc, 0, 0, w3, FFc, 0, 0, ff, FFc, 0, 0,
              nullptr, BWc, FFc, Cc, 1.f, 0, /*relu^2*/2, 0, 0, 1, 1);
  launch_gemm(stream, ff, FFc, 0, 0, w4, Cc, 0, 0, x1, Cc, 0, 0,
              nullptr, BWc, Cc, FFc, 1.f, 1, 0, 0, 0, 1, 1);

  // out[b,0,f,w] = x1[b,w,f]
  transpose_out_kernel<<<gEl, 256, 0, stream>>>(x1, out, tot);
}
